// Attention_6116033429933
// MI455X (gfx1250) — matrix operand order repaired and checked by reading, not compiled, not run on hardware
//
#include <hip/hip_runtime.h>
#include <hip/hip_bf16.h>

typedef __attribute__((ext_vector_type(16))) _Float16 v16h;
typedef __attribute__((ext_vector_type(8)))  _Float16 v8h;
typedef __attribute__((ext_vector_type(8)))  float    v8f;
typedef __attribute__((ext_vector_type(4)))  unsigned int v4u;
typedef __attribute__((ext_vector_type(8)))  int v8i;
typedef __attribute__((ext_vector_type(4)))  int v4i;

static __device__ __forceinline__ v8f zero8() {
  v8f c;
#pragma unroll
  for (int i = 0; i < 8; ++i) c[i] = 0.0f;
  return c;
}

static __device__ __forceinline__ v16h cat16(v8h lo, v8h hi) {
  return __builtin_shufflevector(lo, hi, 0,1,2,3,4,5,6,7,8,9,10,11,12,13,14,15);
}

// A-fragment: 16x32 f16, row-major source with leading dim ldk (halves).
// lane<16: row=lane, K {0..7,16..23}; lane>=16: row=lane-16, K {8..15,24..31}
static __device__ __forceinline__ v16h load_a_frag(const _Float16* __restrict__ base,
                                                   int ldk, int lane) {
  const int hf = lane >> 4;
  const int r  = lane & 15;
  const _Float16* p = base + r * ldk + hf * 8;
  return cat16(*(const v8h*)(p), *(const v8h*)(p + 16));
}

// B-fragment: 32x16 f16 from a TRANSPOSED source Bt[n][k] (k contiguous, ld=ldk).
// lane<16: col=lane, K {0..7,16..23}; lane>=16: col=lane-16, K {8..15,24..31}
static __device__ __forceinline__ v16h load_b_frag(const _Float16* __restrict__ base,
                                                   int ldk, int lane) {
  const int hf = lane >> 4;
  const int n  = lane & 15;
  const _Float16* p = base + n * ldk + hf * 8;
  return cat16(*(const v8h*)(p), *(const v8h*)(p + 16));
}

static __device__ __forceinline__ v8f wmma_f16(v16h a, v16h b, v8f c) {
  return __builtin_amdgcn_wmma_f32_16x16x32_f16(false, a, false, b, (short)0, c,
                                                false, false);
}

// ---------------------------------------------------------------------------
// TDM: issue a 2D f16 tile load Global -> LDS via the Tensor Data Mover.
// D# layout per CDNA5 ISA ch.8 (group0 128b, group1 256b; groups 2/3 zero for
// 2D tiles). data_size code 1 = 2 bytes. Tracked with TENSORcnt.
// 6-arg builtin form (clang-23 / therock-10.0 headers).
// ---------------------------------------------------------------------------
static __device__ __forceinline__ void tdm_load_2d_f16(
    unsigned int lds_byte_addr, const void* global_tile_start,
    unsigned int tensor_d0, unsigned int tensor_d1,  // elements
    unsigned int tile_d0, unsigned int tile_d1,      // elements
    unsigned int d0_stride)                          // elements
{
  const unsigned long long ga = (unsigned long long)global_tile_start;
  v4u g0;
  g0[0] = 1u;                                          // count=1, user descriptor
  g0[1] = lds_byte_addr;                               // lds_addr
  g0[2] = (unsigned int)(ga & 0xffffffffu);            // global_addr[31:0]
  g0[3] = (unsigned int)((ga >> 32) & 0x01ffffffu)     // global_addr[56:32]
          | (2u << 30);                                // type=2 ("image")
  v8i g1;
  g1[0] = (int)(1u << 16);                             // data_size=1 (2B)
  g1[1] = (int)((tensor_d0 & 0xffffu) << 16);          // tensor_dim0[15:0]
  g1[2] = (int)((tensor_d0 >> 16) | ((tensor_d1 & 0xffffu) << 16));
  g1[3] = (int)((tensor_d1 >> 16) | (tile_d0 << 16));  // tile_dim0
  g1[4] = (int)(tile_d1 & 0xffffu);                    // tile_dim1 (tile_dim2=0)
  g1[5] = (int)d0_stride;                              // tensor_dim0_stride[31:0]
  g1[6] = 0;                                           // stride[47:32] | dim1_stride lo
  g1[7] = 0;
  const v4i z4 = {0, 0, 0, 0};
  const v8i z8 = {0, 0, 0, 0, 0, 0, 0, 0};
  __builtin_amdgcn_tensor_load_to_lds(g0, g1, z4, z4, z8, 0);
}

static __device__ __forceinline__ unsigned int lds_addr_of(const void* p) {
  return (unsigned int)(unsigned long long)(uintptr_t)p;  // low 32b = LDS offset
}

// ---------------------------------------------------------------------------
// 1) RMSNorm: x[8192][1024] f32 -> xn f16
// ---------------------------------------------------------------------------
__global__ __launch_bounds__(256) void rmsnorm_kernel(const float* __restrict__ x,
                                                      const float* __restrict__ scale,
                                                      _Float16* __restrict__ xn) {
  __shared__ float red[256];
  const int row = blockIdx.x;
  const int tid = threadIdx.x;
  const float4 v = ((const float4*)(x + row * 1024))[tid];
  red[tid] = v.x * v.x + v.y * v.y + v.z * v.z + v.w * v.w;
  __syncthreads();
#pragma unroll
  for (int s = 128; s > 0; s >>= 1) {
    if (tid < s) red[tid] += red[tid + s];
    __syncthreads();
  }
  const float rr = rsqrtf(red[0] * (1.0f / 1024.0f) + 1e-6f);
  const float4 s4 = ((const float4*)scale)[tid];
  _Float16* o = xn + row * 1024 + tid * 4;
  o[0] = (_Float16)(v.x * rr * s4.x);
  o[1] = (_Float16)(v.y * rr * s4.y);
  o[2] = (_Float16)(v.z * rr * s4.z);
  o[3] = (_Float16)(v.w * rr * s4.w);
}

// ---------------------------------------------------------------------------
// 2) Weight convert + transpose: W[K][N] f32 -> Wt[N][K] f16
// ---------------------------------------------------------------------------
__global__ void transpose_f16_kernel(const float* __restrict__ W,
                                     _Float16* __restrict__ Wt, int K, int N) {
  const int idx = blockIdx.x * blockDim.x + threadIdx.x;
  if (idx >= K * N) return;
  const int k = idx / N;
  const int n = idx - k * N;
  Wt[n * K + k] = (_Float16)W[idx];
}

// ---------------------------------------------------------------------------
// TDM-fed, double-buffered WMMA GEMM core (shared by QKV and proj kernels).
// Block = 256 threads (8 waves), block tile = 64(M) x 128(N), K-step = 32.
// Wave w computes the 32x32 sub-tile at (w>>2)*32, (w&3)*32.
// ---------------------------------------------------------------------------
#define GEMM_TDM_MAINLOOP(A_PTR, BT_PTR, A_ROWS, BT_ROWS)                        \
  __shared__ __align__(16) _Float16 sA[2][64 * 32];                              \
  __shared__ __align__(16) _Float16 sB[2][128 * 32];                             \
  const int lane = threadIdx.x & 31;                                             \
  const int wave = threadIdx.x >> 5;                                             \
  const int m_off = (wave >> 2) * 32;                                            \
  const int n_off = (wave & 3) * 32;                                             \
  const int hf = lane >> 4;                                                      \
  const int r16 = lane & 15;                                                     \
  const int m0blk = blockIdx.x * 64;                                             \
  const int n0blk = blockIdx.y * 128;                                            \
  v8f c00 = zero8(), c01 = zero8(), c10 = zero8(), c11 = zero8();                \
  if (wave == 0) {                                                               \
    tdm_load_2d_f16(lds_addr_of(&sA[0][0]), (A_PTR) + m0blk * 1024,              \
                    1024u, (A_ROWS), 32u, 64u, 1024u);                           \
    tdm_load_2d_f16(lds_addr_of(&sB[0][0]), (BT_PTR) + n0blk * 1024,             \
                    1024u, (BT_ROWS), 32u, 128u, 1024u);                         \
  }                                                                              \
  for (int it = 0; it < 32; ++it) {                                              \
    const int cur = it & 1;                                                      \
    const int nxt = cur ^ 1;                                                     \
    if (wave == 0) {                                                             \
      if (it + 1 < 32) {                                                         \
        const int k0 = (it + 1) * 32;                                            \
        tdm_load_2d_f16(lds_addr_of(&sA[nxt][0]),                                \
                        (A_PTR) + m0blk * 1024 + k0, 1024u, (A_ROWS), 32u, 64u,  \
                        1024u);                                                  \
        tdm_load_2d_f16(lds_addr_of(&sB[nxt][0]),                                \
                        (BT_PTR) + n0blk * 1024 + k0, 1024u, (BT_ROWS), 32u,     \
                        128u, 1024u);                                            \
        __builtin_amdgcn_s_wait_tensorcnt(2); /* current pair retired */         \
      } else {                                                                   \
        __builtin_amdgcn_s_wait_tensorcnt(0);                                    \
      }                                                                          \
    }                                                                            \
    __syncthreads();                                                             \
    {                                                                            \
      const _Float16* pa0 = &sA[cur][(m_off + 0 + r16) * 32 + hf * 8];           \
      const _Float16* pa1 = &sA[cur][(m_off + 16 + r16) * 32 + hf * 8];          \
      const _Float16* pb0 = &sB[cur][(n_off + 0 + r16) * 32 + hf * 8];           \
      const _Float16* pb1 = &sB[cur][(n_off + 16 + r16) * 32 + hf * 8];          \
      const v16h a0 = cat16(*(const v8h*)(pa0), *(const v8h*)(pa0 + 16));        \
      const v16h a1 = cat16(*(const v8h*)(pa1), *(const v8h*)(pa1 + 16));        \
      const v16h b0 = cat16(*(const v8h*)(pb0), *(const v8h*)(pb0 + 16));        \
      const v16h b1 = cat16(*(const v8h*)(pb1), *(const v8h*)(pb1 + 16));        \
      c00 = wmma_f16(a0, b0, c00);                                               \
      c01 = wmma_f16(a0, b1, c01);                                               \
      c10 = wmma_f16(a1, b0, c10);                                               \
      c11 = wmma_f16(a1, b1, c11);                                               \
    }                                                                            \
    __syncthreads();                                                             \
  }

// ---------------------------------------------------------------------------
// 3) QKV GEMM: xn[8192][1024] @ w_qkv -> scatter f16 into Q,K (pre-RoPE), Vt
// ---------------------------------------------------------------------------
__global__ __launch_bounds__(256) void gemm_qkv_kernel(
    const _Float16* __restrict__ A, const _Float16* __restrict__ Bt,
    _Float16* __restrict__ Qb, _Float16* __restrict__ Kb,
    _Float16* __restrict__ Vt) {
  GEMM_TDM_MAINLOOP(A, Bt, 8192u, 3072u)

  const int m0 = m0blk + m_off;
  const int n0 = n0blk + n_off;
  auto emit = [&](const v8f& c, int mi, int ni) {
#pragma unroll
    for (int r = 0; r < 8; ++r) {
      const int row = m0 + mi + r + hf * 8;  // 0..8191
      const int col = n0 + ni + r16;         // 0..3071
      const int s = col >> 10;
      const int c2 = col & 1023;
      const int h = c2 >> 6;
      const int d = c2 & 63;
      const int b = row >> 10;
      const int n = row & 1023;
      const int bh = b * 16 + h;
      const _Float16 hv = (_Float16)c[r];
      if (s == 0)       Qb[(bh * 1024 + n) * 64 + d] = hv;
      else if (s == 1)  Kb[(bh * 1024 + n) * 64 + d] = hv;
      else              Vt[(bh * 64 + d) * 1024 + n] = hv;
    }
  };
  emit(c00, 0, 0); emit(c01, 0, 16); emit(c10, 16, 0); emit(c11, 16, 16);
}

// ---------------------------------------------------------------------------
// 4) 2D RoPE applied in place to Q and K (f16). One thread per rotation pair.
// ---------------------------------------------------------------------------
__global__ void rope_kernel(_Float16* __restrict__ Qb, _Float16* __restrict__ Kb) {
  const int idx = blockIdx.x * blockDim.x + threadIdx.x;  // [0, 8388608)
  _Float16* buf = (idx < 4194304) ? Qb : Kb;
  const int t = idx & 4194303;
  const int j = t & 31;            // pair index 0..31
  const int n = (t >> 5) & 1023;   // token position
  const int bh = t >> 15;          // 0..127
  _Float16* p = buf + (bh * 1024 + n) * 64 + 2 * j;
  const float x1 = (float)p[0];
  const float x2 = (float)p[1];
  const float fi = (float)(j & 15) * (1.0f / 16.0f);
  const float freq = __powf(10000.0f, -fi);
  const float pos = (j < 16) ? (float)(n & 31) : (float)(n >> 5);
  float s, c;
  __sincosf(pos * freq, &s, &c);
  p[0] = (_Float16)(x1 * c - x2 * s);
  p[1] = (_Float16)(x1 * s + x2 * c);
}

// ---------------------------------------------------------------------------
// 5) Flash attention. Block = 128 threads (4 waves). Each wave: 16 query rows.
// ---------------------------------------------------------------------------
__global__ __launch_bounds__(128) void attn_kernel(
    const _Float16* __restrict__ Qb, const _Float16* __restrict__ Kb,
    const _Float16* __restrict__ Vt, _Float16* __restrict__ Ob) {
  __shared__ __align__(16) _Float16 sP[4 * 16 * 32];  // per-wave 16x32 P tile
  const int lane = threadIdx.x & 31;
  const int wave = threadIdx.x >> 5;
  const int hf = lane >> 4;
  const int r16 = lane & 15;
  const int bh = blockIdx.x >> 4;
  const int q0 = (blockIdx.x & 15) * 64 + wave * 16;
  const int b = bh >> 4;
  const int h = bh & 15;

  const _Float16* qbase = Qb + (bh * 1024 + q0) * 64;
  const v16h qa0 = load_a_frag(qbase, 64, lane);       // d 0..31
  const v16h qa1 = load_a_frag(qbase + 32, 64, lane);  // d 32..63

  v8f o0 = zero8(), o1 = zero8(), o2 = zero8(), o3 = zero8();
  float m[8], l[8];
#pragma unroll
  for (int r = 0; r < 8; ++r) { m[r] = -1e30f; l[r] = 0.0f; }
  _Float16* sp = sP + wave * 512;

  for (int kt = 0; kt < 1024; kt += 32) {
    const _Float16* kbase = Kb + (bh * 1024 + kt) * 64;
    v8f s0 = zero8(), s1 = zero8();
    s0 = wmma_f16(qa0, load_b_frag(kbase, 64, lane), s0);
    s0 = wmma_f16(qa1, load_b_frag(kbase + 32, 64, lane), s0);
    s1 = wmma_f16(qa0, load_b_frag(kbase + 16 * 64, 64, lane), s1);
    s1 = wmma_f16(qa1, load_b_frag(kbase + 16 * 64 + 32, 64, lane), s1);

#pragma unroll
    for (int r = 0; r < 8; ++r) {
      const float a0 = s0[r] * 0.125f;  // 1/sqrt(64)
      const float a1 = s1[r] * 0.125f;
      float mr = fmaxf(a0, a1);
#pragma unroll
      for (int off = 1; off < 16; off <<= 1)
        mr = fmaxf(mr, __shfl_xor(mr, off, 32));
      const float mn = fmaxf(m[r], mr);
      const float alpha = __expf(m[r] - mn);
      const float p0 = __expf(a0 - mn);
      const float p1 = __expf(a1 - mn);
      float rs = p0 + p1;
#pragma unroll
      for (int off = 1; off < 16; off <<= 1) rs += __shfl_xor(rs, off, 32);
      l[r] = l[r] * alpha + rs;
      m[r] = mn;
      o0[r] *= alpha; o1[r] *= alpha; o2[r] *= alpha; o3[r] *= alpha;
      const int rowoff = (r + hf * 8) * 32 + r16;  // C-layout -> [16][32] tile
      sp[rowoff] = (_Float16)p0;
      sp[rowoff + 16] = (_Float16)p1;
    }

    // Re-fragment P (16x32) into A-layout from LDS (same wave; DScnt ordering).
    const _Float16* pp = sp + r16 * 32 + hf * 8;
    const v16h pa = cat16(*(const v8h*)(pp), *(const v8h*)(pp + 16));

    const _Float16* vbase = Vt + (bh * 64) * 1024 + kt;
    o0 = wmma_f16(pa, load_b_frag(vbase + 0 * 16 * 1024, 1024, lane), o0);
    o1 = wmma_f16(pa, load_b_frag(vbase + 1 * 16 * 1024, 1024, lane), o1);
    o2 = wmma_f16(pa, load_b_frag(vbase + 2 * 16 * 1024, 1024, lane), o2);
    o3 = wmma_f16(pa, load_b_frag(vbase + 3 * 16 * 1024, 1024, lane), o3);
  }

#pragma unroll
  for (int r = 0; r < 8; ++r) {
    const float inv = 1.0f / l[r];
    const int nrow = q0 + r + hf * 8;
    _Float16* ob = Ob + (b * 1024 + nrow) * 1024 + h * 64 + r16;
    ob[0]  = (_Float16)(o0[r] * inv);
    ob[16] = (_Float16)(o1[r] * inv);
    ob[32] = (_Float16)(o2[r] * inv);
    ob[48] = (_Float16)(o3[r] * inv);
  }
}

// ---------------------------------------------------------------------------
// 6) Output projection: Ob[8192][1024] f16 @ w_proj + b_proj -> out f32
// ---------------------------------------------------------------------------
__global__ __launch_bounds__(256) void gemm_proj_kernel(
    const _Float16* __restrict__ A, const _Float16* __restrict__ Bt,
    const float* __restrict__ bias, float* __restrict__ out) {
  GEMM_TDM_MAINLOOP(A, Bt, 8192u, 1024u)

  const int m0 = m0blk + m_off;
  const int n0 = n0blk + n_off;
  auto emit = [&](const v8f& c, int mi, int ni) {
#pragma unroll
    for (int r = 0; r < 8; ++r) {
      const int row = m0 + mi + r + hf * 8;
      const int col = n0 + ni + r16;
      out[row * 1024 + col] = c[r] + bias[col];
    }
  };
  emit(c00, 0, 0); emit(c01, 0, 16); emit(c10, 16, 0); emit(c11, 16, 16);
}

// ---------------------------------------------------------------------------
// Host launcher
// ---------------------------------------------------------------------------
extern "C" void kernel_launch(void* const* d_in, const int* in_sizes, int n_in,
                              void* d_out, int out_size, void* d_ws, size_t ws_size,
                              hipStream_t stream) {
  (void)in_sizes; (void)n_in; (void)out_size; (void)ws_size;
  const float* x      = (const float*)d_in[0];  // [8,1024,1024]
  const float* scale  = (const float*)d_in[1];  // [1024]
  const float* w_qkv  = (const float*)d_in[2];  // [1024,3072]
  const float* w_proj = (const float*)d_in[3];  // [1024,1024]
  const float* b_proj = (const float*)d_in[4];  // [1024]
  float* out = (float*)d_out;

  char* ws = (char*)d_ws;
  _Float16* xn     = (_Float16*)(ws);                            // 16 MiB
  _Float16* wqkvT  = (_Float16*)(ws + (size_t)(16) * (1 << 20)); //  6 MiB
  _Float16* wprojT = (_Float16*)(ws + (size_t)(22) * (1 << 20)); //  2 MiB
  _Float16* Qb     = (_Float16*)(ws + (size_t)(24) * (1 << 20)); // 16 MiB
  _Float16* Kb     = (_Float16*)(ws + (size_t)(40) * (1 << 20)); // 16 MiB
  _Float16* Vt     = (_Float16*)(ws + (size_t)(56) * (1 << 20)); // 16 MiB
  _Float16* Ob     = (_Float16*)(ws + (size_t)(72) * (1 << 20)); // 16 MiB

  rmsnorm_kernel<<<8192, 256, 0, stream>>>(x, scale, xn);
  transpose_f16_kernel<<<(1024 * 3072) / 256, 256, 0, stream>>>(w_qkv, wqkvT, 1024, 3072);
  transpose_f16_kernel<<<(1024 * 1024) / 256, 256, 0, stream>>>(w_proj, wprojT, 1024, 1024);
  gemm_qkv_kernel<<<dim3(128, 24), 256, 0, stream>>>(xn, wqkvT, Qb, Kb, Vt);
  rope_kernel<<<(2 * 4194304) / 256, 256, 0, stream>>>(Qb, Kb);
  attn_kernel<<<2048, 128, 0, stream>>>(Qb, Kb, Vt, Ob);
  gemm_proj_kernel<<<dim3(128, 8), 256, 0, stream>>>(Ob, wprojT, b_proj, out);
}
